// ConvAttnLayer_71012989272752
// MI455X (gfx1250) — matrix operand order repaired and checked by reading, not compiled, not run on hardware
//
#include <hip/hip_runtime.h>

#define B_ 16
#define L_ 512
#define J_ 4
#define O_ 4
#define D_ 64

typedef __attribute__((ext_vector_type(16))) __bf16 v16bf;
typedef __attribute__((ext_vector_type(8)))  float  v8f;

union AB { v16bf v; unsigned u[8]; };
union CD { v8f v; float f[8]; };

// round-to-nearest-even f32 -> bf16 helpers
__device__ inline unsigned pk2bf(float a, float b) {
    unsigned ua = __float_as_uint(a), ub = __float_as_uint(b);
    ua = (ua + 0x7FFFu + ((ua >> 16) & 1u)) >> 16;
    ub = (ub + 0x7FFFu + ((ub >> 16) & 1u)) >> 16;
    return (ua & 0xFFFFu) | (ub << 16);
}
__device__ inline unsigned short f2bf(float a) {
    unsigned ua = __float_as_uint(a);
    ua = (ua + 0x7FFFu + ((ua >> 16) & 1u)) >> 16;
    return (unsigned short)ua;
}

// A-fragment K index for a given (vgpr v, half-wave h), bf16 16x16x32 layout
#define AK0(v, h) (((v) < 4) ? (8 * (h) + 2 * (v)) : (16 + 8 * (h) + 2 * ((v) - 4)))

#define XROW 33    // padded row stride (uints) for Xs / Wt
#define PROW 258   // padded row stride (uints) for per-wave P buffer (even: b64-aligned pairs)

__global__ __launch_bounds__(128, 1)
void conv_attn_kernel(const float* __restrict__ inp, const float* __restrict__ Wg_,
                      float* __restrict__ out, float* __restrict__ attn) {
    __shared__ unsigned Xs[512 * XROW];        // X_j[b] bf16 pairs along e   (~67.6 KB)
    __shared__ unsigned Wt[64 * XROW];         // W^T  bf16 pairs along d     (~8.4 KB)
    __shared__ unsigned PE[4 * 16 * PROW];     // per-wave 16x512 bf16 P      (~66.0 KB)
    __shared__ float    invS[4 * 16];          // per-wave per-row 1/Z

    const int tid  = threadIdx.x;
    const int wave = tid >> 5;
    const int lane = tid & 31;
    const int n16  = lane & 15;
    const int h    = lane >> 4;

    const int rb = blockIdx.x & 7;            // 64-row block (8 per L)
    const int b  = (blockIdx.x >> 3) & 15;
    const int o  = blockIdx.x >> 7;

    const int row0 = rb * 64 + wave * 16;     // wave's first global row

    unsigned*       PEw   = &PE[wave * 16 * PROW];
    unsigned short* PEw16 = (unsigned short*)PEw;
    unsigned short* Xs16  = (unsigned short*)Xs;

    CD ctx[4];
    #pragma unroll
    for (int nt = 0; nt < 4; ++nt)
        #pragma unroll
        for (int r = 0; r < 8; ++r) ctx[nt].f[r] = 0.f;

    #pragma unroll 1
    for (int j = 0; j < J_; ++j) {
        __syncthreads();  // protect Xs/Wt against reuse from previous j

        // ---- cooperative load X_j[b] -> Xs, W[o][j] -> Wt (f32 -> bf16)
        {
            const float* Xg = inp + (size_t)b * L_ * J_ * D_ + (size_t)j * D_;
            #pragma unroll 2
            for (int it = 0; it < 64; ++it) {
                int idx = tid + it * 128;        // float4 index 0..8191
                int m   = idx >> 4;              // row 0..511
                int e4  = (idx & 15) << 2;       // e 0..60 step 4
                const float4 x = *(const float4*)(Xg + (size_t)m * (J_ * D_) + e4);
                Xs[m * XROW + (e4 >> 1) + 0] = pk2bf(x.x, x.y);
                Xs[m * XROW + (e4 >> 1) + 1] = pk2bf(x.z, x.w);
            }
            const float* Wg = Wg_ + (size_t)(o * J_ + j) * D_ * D_;
            #pragma unroll 2
            for (int it = 0; it < 16; ++it) {
                int u  = tid + it * 128;         // 0..2047
                int dp = u >> 6;                 // d-pair 0..31
                int e  = u & 63;
                Wt[e * XROW + dp] = pk2bf(Wg[(2 * dp) * D_ + e], Wg[(2 * dp + 1) * D_ + e]);
            }
            if (j < J_ - 1) {  // hint next panel into cache: global_prefetch_b8
                const char* nxt = (const char*)(inp + (size_t)b * L_ * J_ * D_ + (size_t)(j + 1) * D_);
                __builtin_prefetch(nxt + (size_t)tid * 4 * (J_ * D_) * 4, 0, 0);
            }
        }
        __syncthreads();

        // ---- matmul 1: q(16x64) = X_rows @ W ; stage q/8 as bf16 into PEw
        {
            AB aX[2];
            #pragma unroll
            for (int ks = 0; ks < 2; ++ks)
                #pragma unroll
                for (int v = 0; v < 8; ++v)
                    aX[ks].u[v] = Xs[(row0 + n16) * XROW + ((AK0(v, h) + 32 * ks) >> 1)];
            #pragma unroll
            for (int nt = 0; nt < 4; ++nt) {
                AB bW[2];
                #pragma unroll
                for (int ks = 0; ks < 2; ++ks)
                    #pragma unroll
                    for (int v = 0; v < 8; ++v)
                        bW[ks].u[v] = Wt[(nt * 16 + n16) * XROW + ((AK0(v, h) + 32 * ks) >> 1)];
                v8f acc = {};
                acc = __builtin_amdgcn_wmma_f32_16x16x32_bf16(false, aX[0].v, false, bW[0].v,
                                                              (short)0, acc, false, false);
                acc = __builtin_amdgcn_wmma_f32_16x16x32_bf16(false, aX[1].v, false, bW[1].v,
                                                              (short)0, acc, false, false);
                CD qc; qc.v = acc;
                #pragma unroll
                for (int r = 0; r < 8; ++r) {
                    int e = nt * 16 + n16;
                    PEw16[2 * ((r + 8 * h) * PROW + (e >> 1)) + (e & 1)] = f2bf(qc.f[r] * 0.125f);
                }
            }
        }
        asm volatile("s_wait_dscnt 0x0" ::: "memory");

        // A-fragments of q, kept in registers for both S passes
        AB aQ[2];
        #pragma unroll
        for (int ks = 0; ks < 2; ++ks)
            #pragma unroll
            for (int v = 0; v < 8; ++v)
                aQ[ks].u[v] = PEw[n16 * PROW + ((AK0(v, h) + 32 * ks) >> 1)];

        // ---- pass A: stream S tiles, track row max (S discarded)
        float mx[8];
        #pragma unroll
        for (int r = 0; r < 8; ++r) mx[r] = -3.0e38f;
        #pragma unroll 1
        for (int ct = 0; ct < 32; ++ct) {
            AB b0, b1;
            #pragma unroll
            for (int v = 0; v < 8; ++v) b0.u[v] = Xs[(ct * 16 + n16) * XROW + (AK0(v, h) >> 1)];
            #pragma unroll
            for (int v = 0; v < 8; ++v) b1.u[v] = Xs[(ct * 16 + n16) * XROW + ((AK0(v, h) + 32) >> 1)];
            v8f acc = {};
            acc = __builtin_amdgcn_wmma_f32_16x16x32_bf16(false, aQ[0].v, false, b0.v,
                                                          (short)0, acc, false, false);
            acc = __builtin_amdgcn_wmma_f32_16x16x32_bf16(false, aQ[1].v, false, b1.v,
                                                          (short)0, acc, false, false);
            CD c; c.v = acc;
            #pragma unroll
            for (int r = 0; r < 8; ++r) mx[r] = fmaxf(mx[r], c.f[r]);
        }
        #pragma unroll
        for (int dd = 1; dd < 16; dd <<= 1)
            #pragma unroll
            for (int r = 0; r < 8; ++r) mx[r] = fmaxf(mx[r], __shfl_xor(mx[r], dd, 16));

        // ---- pass B: recompute S, exp in f32, row-sum, store bf16 exp into PEw
        float sm[8];
        #pragma unroll
        for (int r = 0; r < 8; ++r) sm[r] = 0.f;
        #pragma unroll 1
        for (int ct = 0; ct < 32; ++ct) {
            AB b0, b1;
            #pragma unroll
            for (int v = 0; v < 8; ++v) b0.u[v] = Xs[(ct * 16 + n16) * XROW + (AK0(v, h) >> 1)];
            #pragma unroll
            for (int v = 0; v < 8; ++v) b1.u[v] = Xs[(ct * 16 + n16) * XROW + ((AK0(v, h) + 32) >> 1)];
            v8f acc = {};
            acc = __builtin_amdgcn_wmma_f32_16x16x32_bf16(false, aQ[0].v, false, b0.v,
                                                          (short)0, acc, false, false);
            acc = __builtin_amdgcn_wmma_f32_16x16x32_bf16(false, aQ[1].v, false, b1.v,
                                                          (short)0, acc, false, false);
            CD c; c.v = acc;
            int col = ct * 16 + n16;
            #pragma unroll
            for (int r = 0; r < 8; ++r) {
                float e = __expf(c.f[r] - mx[r]);
                sm[r] += e;
                PEw16[2 * ((r + 8 * h) * PROW + (col >> 1)) + (col & 1)] = f2bf(e);
            }
        }
        #pragma unroll
        for (int dd = 1; dd < 16; dd <<= 1)
            #pragma unroll
            for (int r = 0; r < 8; ++r) sm[r] += __shfl_xor(sm[r], dd, 16);
        float inv[8];
        #pragma unroll
        for (int r = 0; r < 8; ++r) inv[r] = 1.0f / sm[r];
        if (n16 == 0) {
            #pragma unroll
            for (int r = 0; r < 8; ++r) invS[wave * 16 + 8 * h + r] = inv[r];
        }
        asm volatile("s_wait_dscnt 0x0" ::: "memory");

        // ---- attn output: b64 LDS reads, normalize, b128 (float4) coalesced stores
        {
            float* attnW = attn + (((size_t)(o * J_ + j) * B_ + b) * L_ + row0) * L_;
            #pragma unroll 1
            for (int row = 0; row < 16; ++row) {
                float iv = invS[wave * 16 + row];
                #pragma unroll 1
                for (int it = 0; it < 4; ++it) {
                    int idx2 = it * 64 + 2 * lane;         // even uint column 0..254
                    uint2 u2 = *(const uint2*)&PEw[row * PROW + idx2];
                    float4 val;
                    val.x = __uint_as_float(u2.x << 16) * iv;
                    val.y = __uint_as_float(u2.x & 0xFFFF0000u) * iv;
                    val.z = __uint_as_float(u2.y << 16) * iv;
                    val.w = __uint_as_float(u2.y & 0xFFFF0000u) * iv;
                    *(float4*)(attnW + (size_t)row * L_ + 2 * idx2) = val;
                }
            }
        }

        // ---- matmul 3: ctxj(16x64) = expS(16x512) @ X(512x64); normalize into ctx
        {
            CD ctxj[4];
            #pragma unroll
            for (int nt = 0; nt < 4; ++nt)
                #pragma unroll
                for (int r = 0; r < 8; ++r) ctxj[nt].f[r] = 0.f;
            #pragma unroll 1
            for (int ksl = 0; ksl < 16; ++ksl) {
                AB aP;
                #pragma unroll
                for (int v = 0; v < 8; ++v)
                    aP.u[v] = PEw[n16 * PROW + ((AK0(v, h) + 32 * ksl) >> 1)];
                int kbase = ksl * 32;
                #pragma unroll
                for (int nt = 0; nt < 4; ++nt) {
                    AB bX;
                    #pragma unroll
                    for (int v = 0; v < 8; ++v) {
                        int krow = kbase + AK0(v, h);        // X row (pairs along K=m)
                        int dcol = nt * 16 + n16;            // X col d
                        unsigned lo = Xs16[2 * (krow * XROW + (dcol >> 1)) + (dcol & 1)];
                        unsigned hi = Xs16[2 * ((krow + 1) * XROW + (dcol >> 1)) + (dcol & 1)];
                        bX.u[v] = lo | (hi << 16);
                    }
                    ctxj[nt].v = __builtin_amdgcn_wmma_f32_16x16x32_bf16(false, aP.v, false, bX.v,
                                                                         (short)0, ctxj[nt].v, false, false);
                }
            }
            #pragma unroll
            for (int nt = 0; nt < 4; ++nt)
                #pragma unroll
                for (int r = 0; r < 8; ++r) ctx[nt].f[r] += ctxj[nt].f[r] * inv[r];
        }
    }  // j loop

    // ---- out = mean over j
    const float outScale = 1.0f / (float)J_;
    #pragma unroll
    for (int nt = 0; nt < 4; ++nt)
        #pragma unroll
        for (int r = 0; r < 8; ++r) {
            int row  = row0 + r + 8 * h;
            int dcol = nt * 16 + n16;
            out[(((size_t)b * L_ + row) * O_ + o) * D_ + dcol] = ctx[nt].f[r] * outScale;
        }
}

extern "C" void kernel_launch(void* const* d_in, const int* in_sizes, int n_in,
                              void* d_out, int out_size, void* d_ws, size_t ws_size,
                              hipStream_t stream) {
    const float* inp = (const float*)d_in[0];
    const float* W   = (const float*)d_in[1];
    float* out  = (float*)d_out;
    float* attn = out + (size_t)B_ * L_ * O_ * D_;  // outputs concatenated: [out][attn]
    dim3 grid(O_ * B_ * (L_ / 64));  // 512 blocks, 4 waves each
    conv_attn_kernel<<<grid, 128, 0, stream>>>(inp, W, out, attn);
}
